// GraphModel_44032004719052
// MI455X (gfx1250) — compile-verified, matrix operand order checked by reading, not run on hardware
//
#include <hip/hip_runtime.h>

// Problem constants (match reference)
#define BQ   4
#define NQ   8192
#define EQ   131072
#define RQ   16
#define HQ   128
#define OUTQ 32

typedef __attribute__((ext_vector_type(16))) __bf16 v16bf;
typedef __attribute__((ext_vector_type(8)))  __bf16 v8bf;
typedef __attribute__((ext_vector_type(8)))  float  v8f;

__device__ __forceinline__ unsigned short f32_to_bf16(float f) {
  unsigned int u = __float_as_uint(f);
  u += 0x7FFFu + ((u >> 16) & 1u);   // round-to-nearest-even
  return (unsigned short)(u >> 16);
}
__device__ __forceinline__ float bf16u_to_f32(unsigned short h) {
  return __uint_as_float(((unsigned int)h) << 16);
}

// ---------------------------------------------------------------------------
// h0[b,n,:] = bf16(embed[class_objects[b,n], :])
__global__ void k_embed_gather(const int* __restrict__ cls,
                               const float* __restrict__ embed,
                               unsigned short* __restrict__ h) {
  int i = blockIdx.x * blockDim.x + threadIdx.x;   // over B*N*H
  int n = i >> 7;          // node index (B*N)
  int f = i & (HQ - 1);
  int c = cls[n];
  h[i] = f32_to_bf16(embed[(size_t)c * HQ + f]);
}

// Wt[r,o,k] = bf16(W[r,k,o])  (so WMMA B operand loads like A: contiguous K)
__global__ void k_transpose_w(const float* __restrict__ W,
                              unsigned short* __restrict__ Wt,
                              int K, int O) {
  int i = blockIdx.x * blockDim.x + threadIdx.x;   // over R*K*O
  int per = K * O;
  int r = i / per;
  int rem = i - r * per;
  int o = rem / K;
  int k = rem - o * K;
  Wt[i] = f32_to_bf16(W[(size_t)r * per + (size_t)k * O + o]);
}

__global__ void k_zero_f32(float* __restrict__ p) {
  int i = blockIdx.x * blockDim.x + threadIdx.x;
  p[i] = 0.0f;
}

// accum f32 -> relu -> bf16 h (next layer input)
__global__ void k_relu_cvt(const float* __restrict__ acc,
                           unsigned short* __restrict__ h) {
  int i = blockIdx.x * blockDim.x + threadIdx.x;
  h[i] = f32_to_bf16(fmaxf(acc[i], 0.0f));
}

// ---------------------------------------------------------------------------
// Dense projection: proj[n, r, :] = h[n, :] @ W[r]
// One workgroup per (node-tile tM, relation r); wave w computes out-tile tN=w.
// The shared 16x128 A tile is staged once into LDS (4 KiB) and every wave
// reads its per-lane WMMA fragments from LDS -> 8x less A traffic (HO=128).
// K = HQ = 128 -> 4 chained v_wmma_f32_16x16x32_bf16 with f32 accumulation.
template <int HO>
__global__ void k_proj_wmma(const unsigned short* __restrict__ hb,  // [NQ, HQ] bf16 (batch slice)
                            const unsigned short* __restrict__ Wt,  // [RQ, HO, HQ] bf16
                            unsigned short* __restrict__ proj) {    // [NQ, RQ, HO] bf16
  constexpr int NT = HO >> 4;                       // waves per block = out tiles
  __shared__ unsigned short As[16 * HQ];            // 4 KiB A tile

  int tid  = threadIdx.x;
  int lane = tid & 31;
  int wv   = tid >> 5;                              // tN
  int tM   = blockIdx.x & ((NQ >> 4) - 1);
  int r    = blockIdx.x >> (13 - 4);                // blockIdx.x / (NQ/16)

  // Cooperative stage of the A tile (uniform trip count -> no divergence)
  {
    const uint4* src = (const uint4*)(hb + (size_t)(tM << 4) * HQ);
    uint4* dst = (uint4*)As;
    constexpr int CHUNKS = (16 * HQ * 2) / 16;      // 256 x 16B
    for (int c = tid; c < CHUNKS; c += NT * 32) dst[c] = src[c];
  }
  __syncthreads();

  int lrow  = lane & 15;
  int halfo = (lane >> 4) << 3;    // lanes 0-15: K 0..7/16..23; lanes 16-31: K 8..15/24..31

  const __bf16* A_ptr = (const __bf16*)As + lrow * HQ + halfo;
  const __bf16* B_ptr = (const __bf16*)Wt + ((size_t)r * HO + (wv << 4) + lrow) * HQ + halfo;

  v8f acc = {};
#pragma unroll
  for (int k = 0; k < HQ; k += 32) {
    v8bf alo = *(const v8bf*)(A_ptr + k);           // ds_load_b128
    v8bf ahi = *(const v8bf*)(A_ptr + k + 16);
    v8bf blo = *(const v8bf*)(B_ptr + k);           // global_load_b128
    v8bf bhi = *(const v8bf*)(B_ptr + k + 16);
    v16bf Am, Bm;
#pragma unroll
    for (int i = 0; i < 8; ++i) {
      Am[i] = alo[i]; Am[i + 8] = ahi[i];
      Bm[i] = blo[i]; Bm[i + 8] = bhi[i];
    }
    acc = __builtin_amdgcn_wmma_f32_16x16x32_bf16(
        /*neg_a=*/false, Am, /*neg_b=*/false, Bm,
        /*c_mod=*/(short)0, acc, /*reuse_a=*/false, /*reuse_b=*/false);
  }

  // C/D layout: VGPR j -> M = j + 8*(lane>=16), N = lane&15
  int col   = (wv << 4) + lrow;
  int rbase = (tM << 4) + halfo;
#pragma unroll
  for (int j = 0; j < 8; ++j) {
    size_t row = (size_t)(rbase + j);
    proj[(row * RQ + r) * HO + col] = f32_to_bf16(acc[j]);
  }
}

// ---------------------------------------------------------------------------
// Per-edge gather + atomic scatter-add: accum[dst] += proj[src, rel] * mask_e
template <int HO>
__global__ void k_edge_scatter(const unsigned short* __restrict__ proj, // [NQ,RQ,HO]
                               const int* __restrict__ et,   // [EQ,2] (src,dst)
                               const int* __restrict__ ec,   // [EQ]
                               const float* __restrict__ me, // [EQ]
                               float* __restrict__ accum) {  // [NQ,HO]
  int lane = threadIdx.x & 31;
  int e = blockIdx.x * (blockDim.x >> 5) + (threadIdx.x >> 5);
  int src = et[2 * e + 0];
  int dst = et[2 * e + 1];
  int rel = ec[e];
  float m = me[e];
  const unsigned short* p = proj + ((size_t)src * RQ + rel) * HO;
  float* a = accum + (size_t)dst * HO;
  if constexpr (HO == 128) {
    const unsigned int* p32 = (const unsigned int*)p;  // 4 bf16 per lane
    unsigned int q0 = p32[lane * 2 + 0];
    unsigned int q1 = p32[lane * 2 + 1];
    float v0 = bf16u_to_f32((unsigned short)(q0 & 0xFFFFu)) * m;
    float v1 = bf16u_to_f32((unsigned short)(q0 >> 16)) * m;
    float v2 = bf16u_to_f32((unsigned short)(q1 & 0xFFFFu)) * m;
    float v3 = bf16u_to_f32((unsigned short)(q1 >> 16)) * m;
    int f = lane * 4;
    atomicAdd(a + f + 0, v0);
    atomicAdd(a + f + 1, v1);
    atomicAdd(a + f + 2, v2);
    atomicAdd(a + f + 3, v3);
  } else {
    atomicAdd(a + lane, bf16u_to_f32(p[lane]) * m);
  }
}

// ---------------------------------------------------------------------------
// softmax over OUT=32 (one wave per node) * mask_object
__global__ void k_softmax_mask(const float* __restrict__ acc,   // [B*N, 32]
                               const float* __restrict__ maskn, // [B*N]
                               float* __restrict__ out) {
  int lane = threadIdx.x & 31;
  int node = blockIdx.x * (blockDim.x >> 5) + (threadIdx.x >> 5);
  float v = acc[(size_t)node * OUTQ + lane];
  float mx = v;
#pragma unroll
  for (int s = 16; s >= 1; s >>= 1) mx = fmaxf(mx, __shfl_xor(mx, s, 32));
  float ex = __expf(v - mx);
  float sum = ex;
#pragma unroll
  for (int s = 16; s >= 1; s >>= 1) sum += __shfl_xor(sum, s, 32);
  out[(size_t)node * OUTQ + lane] = (ex / sum) * maskn[node];
}

// ---------------------------------------------------------------------------
extern "C" void kernel_launch(void* const* d_in, const int* in_sizes, int n_in,
                              void* d_out, int out_size, void* d_ws, size_t ws_size,
                              hipStream_t stream) {
  (void)in_sizes; (void)n_in; (void)out_size; (void)ws_size;
  const int*   class_objects = (const int*)d_in[0];
  // d_in[1] = states_objects (unused by reference)
  const int*   edge_tuples   = (const int*)d_in[2];   // [B,E,2]
  const int*   edge_classes  = (const int*)d_in[3];   // [B,E]
  const float* mask_object   = (const float*)d_in[4]; // [B,N]
  const float* mask_edge     = (const float*)d_in[5]; // [B,E]
  const float* embed         = (const float*)d_in[6]; // [100,H]
  const float* W1            = (const float*)d_in[7]; // [R,H,H]
  const float* W2            = (const float*)d_in[8]; // [R,H,H]
  const float* W3            = (const float*)d_in[9]; // [R,H,OUT]
  float* out = (float*)d_out;

  // Workspace layout (all chunks 256B-aligned by construction)
  char* ws = (char*)d_ws;
  size_t off = 0;
  unsigned short* h    = (unsigned short*)(ws + off); off += (size_t)BQ * NQ * HQ * 2;      // 8 MiB
  unsigned short* Wt1  = (unsigned short*)(ws + off); off += (size_t)RQ * HQ * HQ * 2;      // 512 KiB
  unsigned short* Wt2  = (unsigned short*)(ws + off); off += (size_t)RQ * HQ * HQ * 2;      // 512 KiB
  unsigned short* Wt3  = (unsigned short*)(ws + off); off += (size_t)RQ * OUTQ * HQ * 2;    // 128 KiB
  unsigned short* proj = (unsigned short*)(ws + off); off += (size_t)NQ * RQ * HQ * 2;      // 32 MiB (one batch)
  float*          accum = (float*)(ws + off);         off += (size_t)BQ * NQ * HQ * 4;      // 16 MiB

  const int TPB = 256;
  // Stage 0: embed lookup + weight transposes (bf16)
  k_embed_gather<<<(BQ * NQ * HQ) / TPB, TPB, 0, stream>>>(class_objects, embed, h);
  k_transpose_w<<<(RQ * HQ * HQ) / TPB, TPB, 0, stream>>>(W1, Wt1, HQ, HQ);
  k_transpose_w<<<(RQ * HQ * HQ) / TPB, TPB, 0, stream>>>(W2, Wt2, HQ, HQ);
  k_transpose_w<<<(RQ * HQ * OUTQ) / TPB, TPB, 0, stream>>>(W3, Wt3, HQ, OUTQ);

  const int projBlocks = (NQ / 16) * RQ;              // 8192 blocks = (tM, r)
  const int edgeGrid   = EQ / 8;

  // ---- Layers 1 & 2 (H -> H, relu) ----
  const unsigned short* Wts[2] = {Wt1, Wt2};
  for (int layer = 0; layer < 2; ++layer) {
    k_zero_f32<<<(BQ * NQ * HQ) / TPB, TPB, 0, stream>>>(accum);
    for (int b = 0; b < BQ; ++b) {
      k_proj_wmma<HQ><<<projBlocks, (HQ / 16) * 32, 0, stream>>>(
          h + (size_t)b * NQ * HQ, Wts[layer], proj);
      k_edge_scatter<HQ><<<edgeGrid, TPB, 0, stream>>>(
          proj, edge_tuples + (size_t)b * EQ * 2, edge_classes + (size_t)b * EQ,
          mask_edge + (size_t)b * EQ, accum + (size_t)b * NQ * HQ);
    }
    k_relu_cvt<<<(BQ * NQ * HQ) / TPB, TPB, 0, stream>>>(accum, h);
  }

  // ---- Layer 3 (H -> OUT, softmax * mask) ----
  k_zero_f32<<<(BQ * NQ * OUTQ) / TPB, TPB, 0, stream>>>(accum);
  for (int b = 0; b < BQ; ++b) {
    k_proj_wmma<OUTQ><<<projBlocks, (OUTQ / 16) * 32, 0, stream>>>(
        h + (size_t)b * NQ * HQ, Wt3, proj);
    k_edge_scatter<OUTQ><<<edgeGrid, TPB, 0, stream>>>(
        proj, edge_tuples + (size_t)b * EQ * 2, edge_classes + (size_t)b * EQ,
        mask_edge + (size_t)b * EQ, accum + (size_t)b * NQ * OUTQ);
  }
  k_softmax_mask<<<(BQ * NQ) / 8, TPB, 0, stream>>>(accum, mask_object, out);
}
